// MultimodalSarcasmDetector_56264071578182
// MI455X (gfx1250) — matrix operand order repaired and checked by reading, not compiled, not run on hardware
//
#include <hip/hip_runtime.h>
#include <cstddef>

// ---------------------------------------------------------------------------
// Types for CDNA5 WMMA (wave32, 16x16x32 bf16 -> f32)
// ---------------------------------------------------------------------------
typedef __attribute__((ext_vector_type(16))) __bf16        v16bf;
typedef __attribute__((ext_vector_type(8)))  float         v8f;
typedef __attribute__((ext_vector_type(4)))  unsigned int  u32x4;
typedef __attribute__((ext_vector_type(4)))  float         f32x4;

// f32 -> bf16 (round-half-up), packed pair via v_perm_b32: 3 VALU per 2 elems.
__device__ __forceinline__ unsigned int pack2(float lo, float hi) {
  unsigned int a = __float_as_uint(hi) + 0x8000u;
  unsigned int b = __float_as_uint(lo) + 0x8000u;
  // result = { a[31:16], b[31:16] }
  return __builtin_amdgcn_perm(a, b, 0x07060302u);
}
__device__ __forceinline__ unsigned short f2bf1(float f) {
  return (unsigned short)((__float_as_uint(f) + 0x8000u) >> 16);
}

// ---------------------------------------------------------------------------
// Generic bf16-WMMA GEMM:  C[M,N] = act( alpha * A[M,K] x B^T + bias ) + resid
//   TRANSB=false : B is [N,K] row-major (weight rows; also Q.K^T with K-rows)
//   TRANSB=true  : B is [K,N] row-major (att @ V)
// Batched over grid.z with (b,h) decomposition: z = b*nH + h.
// 256 thr = 8 waves. Tile BM=128, BN=64, BK=32, wave tile 32x32.
// Double-buffered LDS: global loads for tile i+1 overlap WMMAs of tile i,
// one workgroup barrier per K-step.
// ---------------------------------------------------------------------------
template<int ACT, bool TRANSB>
__global__ __launch_bounds__(256)
void gemm_bf16_wmma(const float* __restrict__ A, const float* __restrict__ Bm,
                    const float* __restrict__ bias, const float* __restrict__ resid,
                    float* __restrict__ C,
                    int M, int N, int K, int lda, int ldb, int ldc,
                    int nH, long long sAb, long long sAh, long long sBb, long long sBh,
                    long long sCb, long long sCh, float alpha)
{
  __shared__ unsigned short As[2][128][32];
  __shared__ unsigned short Bs[2][64][32];

  const int z  = blockIdx.z;
  const int bb = z / nH;
  const int hh = z - bb * nH;
  A  += (size_t)bb * sAb + (size_t)hh * sAh;
  Bm += (size_t)bb * sBb + (size_t)hh * sBh;
  C  += (size_t)bb * sCb + (size_t)hh * sCh;

  const int m0   = blockIdx.y * 128;
  const int n0   = blockIdx.x * 64;
  const int t    = threadIdx.x;
  const int lane = t & 31;
  const int w    = t >> 5;
  const int wm   = w & 3;          // 4 waves along M -> 32 rows each
  const int wn   = w >> 2;         // 2 waves along N -> 32 cols each
  const int lr   = lane & 15;
  const int lh   = lane >> 4;      // lane half (0/1)

  v8f acc00 = {0.f,0.f,0.f,0.f,0.f,0.f,0.f,0.f};
  v8f acc01 = acc00, acc10 = acc00, acc11 = acc00;

  // tile-loader thread mappings
  const int arow  = t >> 1;        // 0..127
  const int ahalf = t & 1;         // which 16-float chunk of K
  const int bnrow = t >> 2;        // 0..63   (!TRANSB)
  const int bq4   = t & 3;         // 8-float chunk of K (!TRANSB)
  const int bkk   = t >> 3;        // 0..31   (TRANSB)
  const int bq8   = t & 7;         // 8-col group (TRANSB)

  const float* Abase = A + (size_t)(m0 + arow) * lda + ahalf * 16;
  const float* Bbase;
  bool bValid = true;
  if (!TRANSB) {
    const int gn = n0 + bnrow;
    bValid = (gn < N);
    Bbase = Bm + (size_t)(bValid ? gn : 0) * ldb + bq4 * 8;
  } else {
    Bbase = Bm + (size_t)bkk * ldb + n0;
  }

  const f32x4 fz = {0.f, 0.f, 0.f, 0.f};
  f32x4 ra0 = fz, ra1 = fz, ra2 = fz, ra3 = fz;
  f32x4 rb0 = fz, rb1 = fz;     // !TRANSB staging regs
  float rbs[8];                 // TRANSB staging regs
  #pragma unroll
  for (int j = 0; j < 8; ++j) rbs[j] = 0.f;

  auto gload = [&](int k0) {
    const float* ap = Abase + k0;
    ra0 = *reinterpret_cast<const f32x4*>(ap);
    ra1 = *reinterpret_cast<const f32x4*>(ap + 4);
    ra2 = *reinterpret_cast<const f32x4*>(ap + 8);
    ra3 = *reinterpret_cast<const f32x4*>(ap + 12);
    if (!TRANSB) {
      if (bValid) {
        const float* bp = Bbase + k0;
        rb0 = *reinterpret_cast<const f32x4*>(bp);
        rb1 = *reinterpret_cast<const f32x4*>(bp + 4);
      } else {
        rb0 = fz; rb1 = fz;
      }
    } else {
      const float* bp = Bbase + (size_t)k0 * ldb;
      #pragma unroll
      for (int j = 0; j < 8; ++j) {
        int n = bq8 * 8 + j;
        rbs[j] = (n0 + n < N) ? bp[n] : 0.f;
      }
    }
  };

  auto sstore = [&](int p) {
    unsigned int* dA = reinterpret_cast<unsigned int*>(&As[p][arow][ahalf * 16]);
    dA[0] = pack2(ra0[0], ra0[1]); dA[1] = pack2(ra0[2], ra0[3]);
    dA[2] = pack2(ra1[0], ra1[1]); dA[3] = pack2(ra1[2], ra1[3]);
    dA[4] = pack2(ra2[0], ra2[1]); dA[5] = pack2(ra2[2], ra2[3]);
    dA[6] = pack2(ra3[0], ra3[1]); dA[7] = pack2(ra3[2], ra3[3]);
    if (!TRANSB) {
      unsigned int* dB = reinterpret_cast<unsigned int*>(&Bs[p][bnrow][bq4 * 8]);
      dB[0] = pack2(rb0[0], rb0[1]); dB[1] = pack2(rb0[2], rb0[3]);
      dB[2] = pack2(rb1[0], rb1[1]); dB[3] = pack2(rb1[2], rb1[3]);
    } else {
      #pragma unroll
      for (int j = 0; j < 8; ++j) Bs[p][bq8 * 8 + j][bkk] = f2bf1(rbs[j]);
    }
  };

  // prologue: stage tile 0
  gload(0);
  sstore(0);
  __syncthreads();

  const int nk = K >> 5;
  for (int i = 0; i < nk; ++i) {
    const int cur = i & 1;
    if (i + 1 < nk) gload((i + 1) << 5);   // overlap next-tile VMEM with WMMA

    union Frag { u32x4 q[2]; v16bf v; };
    Frag a0, a1, b0, b1;
    const int lo = lh * 8;   // A frag: lane-half selects K 0-7/16-23 vs 8-15/24-31
    {
      const unsigned short* pr = &As[cur][wm * 32 + lr][0];
      a0.q[0] = *reinterpret_cast<const u32x4*>(pr + lo);
      a0.q[1] = *reinterpret_cast<const u32x4*>(pr + 16 + lo);
      pr = &As[cur][wm * 32 + 16 + lr][0];
      a1.q[0] = *reinterpret_cast<const u32x4*>(pr + lo);
      a1.q[1] = *reinterpret_cast<const u32x4*>(pr + 16 + lo);
    }
    {
      const unsigned short* pc = &Bs[cur][wn * 32 + lr][0] + lh * 8;  // B: K 0-7/16-23 vs 8-15/24-31
      b0.q[0] = reinterpret_cast<const u32x4*>(pc)[0];
      b0.q[1] = reinterpret_cast<const u32x4*>(pc)[2];
      pc = &Bs[cur][wn * 32 + 16 + lr][0] + lh * 8;
      b1.q[0] = reinterpret_cast<const u32x4*>(pc)[0];
      b1.q[1] = reinterpret_cast<const u32x4*>(pc)[2];
    }

    acc00 = __builtin_amdgcn_wmma_f32_16x16x32_bf16(false, a0.v, false, b0.v, (short)0, acc00, false, false);
    acc01 = __builtin_amdgcn_wmma_f32_16x16x32_bf16(false, a0.v, false, b1.v, (short)0, acc01, false, false);
    acc10 = __builtin_amdgcn_wmma_f32_16x16x32_bf16(false, a1.v, false, b0.v, (short)0, acc10, false, false);
    acc11 = __builtin_amdgcn_wmma_f32_16x16x32_bf16(false, a1.v, false, b1.v, (short)0, acc11, false, false);

    if (i + 1 < nk) sstore(cur ^ 1);       // fill the other buffer
    __syncthreads();
  }

  // ---- epilogue: C layout lane l, reg r -> m = r + 8*(l>>4), n = l&15 ----
  auto store_tile = [&](const v8f& acc, int mi, int ni) {
    int col = n0 + wn * 32 + ni * 16 + lr;
    if (col >= N) return;
    int rowBase = m0 + wm * 32 + mi * 16 + lh * 8;
    float bv = (bias != nullptr) ? bias[col] : 0.f;
    #pragma unroll
    for (int r = 0; r < 8; ++r) {
      float v = acc[r] * alpha + bv;
      if (ACT == 1) v = 1.f / (1.f + __expf(-v));                 // sigmoid
      else if (ACT == 2) v = (v > 20.f) ? v : log1pf(__expf(v));  // softplus
      size_t ci = (size_t)(rowBase + r) * ldc + col;
      if (resid != nullptr) v += resid[ci];
      C[ci] = v;
    }
  };
  store_tile(acc00, 0, 0); store_tile(acc01, 0, 1);
  store_tile(acc10, 1, 0); store_tile(acc11, 1, 1);
}

// ---------------------------------------------------------------------------
// Row softmax over 512 columns (attention scores). One block per row.
// ---------------------------------------------------------------------------
__global__ __launch_bounds__(256)
void softmax_rows(float* __restrict__ s)
{
  float* row = s + (size_t)blockIdx.x * 512;
  int t = threadIdx.x;
  float a0 = row[t];
  float a1 = row[t + 256];
  __shared__ float red[256];
  red[t] = fmaxf(a0, a1);
  __syncthreads();
  for (int off = 128; off > 0; off >>= 1) {
    if (t < off) red[t] = fmaxf(red[t], red[t + off]);
    __syncthreads();
  }
  float mx = red[0];
  __syncthreads();
  float e0 = __expf(a0 - mx), e1 = __expf(a1 - mx);
  red[t] = e0 + e1;
  __syncthreads();
  for (int off = 128; off > 0; off >>= 1) {
    if (t < off) red[t] += red[t + off];
    __syncthreads();
  }
  float inv = 1.f / red[0];
  row[t] = e0 * inv;
  row[t + 256] = e1 * inv;
}

// ---------------------------------------------------------------------------
// out = LayerNorm( x [+ gate*c] ) over D=768. One block per token row.
// ---------------------------------------------------------------------------
__global__ __launch_bounds__(256)
void add_gate_ln(const float* __restrict__ x, const float* __restrict__ gate,
                 const float* __restrict__ cc, const float* __restrict__ gamma,
                 const float* __restrict__ beta, float* __restrict__ out)
{
  size_t base = (size_t)blockIdx.x * 768;
  int t = threadIdx.x;
  float v[3];
  float sum = 0.f;
  #pragma unroll
  for (int j = 0; j < 3; ++j) {
    int i = t + j * 256;
    float s = x[base + i];
    if (gate != nullptr) s += gate[base + i] * cc[base + i];
    v[j] = s;
    sum += s;
  }
  __shared__ float red[256];
  red[t] = sum;
  __syncthreads();
  for (int off = 128; off > 0; off >>= 1) {
    if (t < off) red[t] += red[t + off];
    __syncthreads();
  }
  float mean = red[0] * (1.f / 768.f);
  __syncthreads();
  float sq = 0.f;
  #pragma unroll
  for (int j = 0; j < 3; ++j) { float d = v[j] - mean; sq += d * d; }
  red[t] = sq;
  __syncthreads();
  for (int off = 128; off > 0; off >>= 1) {
    if (t < off) red[t] += red[t + off];
    __syncthreads();
  }
  float rstd = rsqrtf(red[0] * (1.f / 768.f) + 1e-5f);
  #pragma unroll
  for (int j = 0; j < 3; ++j) {
    int i = t + j * 256;
    out[base + i] = (v[j] - mean) * rstd * gamma[i] + beta[i];
  }
}

// ---------------------------------------------------------------------------
// Depthwise causal conv (D_CONV=4, left pad 3) + bias + SiLU.
// xz is [B,S,3072], channels 0..1535 are the conv input; u is [B,S,1536].
// ---------------------------------------------------------------------------
__global__ __launch_bounds__(256)
void conv1d_silu(const float* __restrict__ xz, const float* __restrict__ w,
                 const float* __restrict__ b, float* __restrict__ u)
{
  int idx = blockIdx.x * 256 + threadIdx.x;       // < 8*512*1536
  int c  = idx % 1536;
  int s  = (idx / 1536) % 512;
  int bb = idx / (1536 * 512);
  float acc = b[c];
  #pragma unroll
  for (int i = 0; i < 4; ++i) {
    int ss = s - 3 + i;
    if (ss >= 0) acc += xz[((size_t)(bb * 512 + ss)) * 3072 + c] * w[c * 4 + i];
  }
  u[(size_t)idx] = acc / (1.f + __expf(-acc));    // silu
}

// ---------------------------------------------------------------------------
// Mamba selective scan. One thread per (batch, channel); 16 states in regs.
// y = (scan_sum + u*D) * silu(z),  z = xz[..., 1536+c]
// ---------------------------------------------------------------------------
__global__ __launch_bounds__(256)
void mamba_scan(const float* __restrict__ u, const float* __restrict__ dlt,
                const float* __restrict__ Bm, const float* __restrict__ A_log,
                const float* __restrict__ Dp, const float* __restrict__ xz,
                float* __restrict__ y)
{
  int idx = blockIdx.x * 256 + threadIdx.x;       // < 8*1536
  int c  = idx % 1536;
  int bb = idx / 1536;
  float a[16], h[16];
  #pragma unroll
  for (int st = 0; st < 16; ++st) {
    a[st] = -__expf(A_log[c * 16 + st]);
    h[st] = 0.f;
  }
  float dpar = Dp[c];
  for (int s = 0; s < 512; ++s) {
    size_t tk = (size_t)(bb * 512 + s);
    float dt = dlt[tk * 1536 + c];
    float uu = u[tk * 1536 + c];
    float udt = uu * dt;
    float ys = 0.f;
    #pragma unroll
    for (int st = 0; st < 16; ++st) {
      h[st] = h[st] * __expf(dt * a[st]) + udt * Bm[tk * 16 + st];
      ys += h[st];
    }
    float zz = xz[tk * 3072 + 1536 + c];
    y[tk * 1536 + c] = (ys + uu * dpar) * (zz / (1.f + __expf(-zz)));
  }
}

// ---------------------------------------------------------------------------
// Host-side orchestration
// ---------------------------------------------------------------------------
struct MHAP  { const float *Wq,*Wk,*Wv,*Wo,*bq,*bk,*bv,*bo; };
struct MambaP{ const float *in_W,*conv_w,*conv_b,*xp_W,*dt_W,*dt_b,*A_log,*D,*out_W,*ln_g,*ln_b; };

static void run_mha(hipStream_t stream,
                    const float* q, const float* k, const float* v, const MHAP& p,
                    float* Q, float* Kb, float* V, float* Sc, float* O, float* out)
{
  dim3 blk(256);
  dim3 gproj(768 / 64, 4096 / 128, 1);
  gemm_bf16_wmma<0,false><<<gproj, blk, 0, stream>>>(q, p.Wq, p.bq, nullptr, Q,
      4096, 768, 768, 768, 768, 768, 1, 0, 0, 0, 0, 0, 0, 1.f);
  gemm_bf16_wmma<0,false><<<gproj, blk, 0, stream>>>(k, p.Wk, p.bk, nullptr, Kb,
      4096, 768, 768, 768, 768, 768, 1, 0, 0, 0, 0, 0, 0, 1.f);
  gemm_bf16_wmma<0,false><<<gproj, blk, 0, stream>>>(v, p.Wv, p.bv, nullptr, V,
      4096, 768, 768, 768, 768, 768, 1, 0, 0, 0, 0, 0, 0, 1.f);

  // scores[b,h] = (Qh Kh^T) / sqrt(96); per-head GEMM over grid.z = 64
  const float scl = 0.1020620726159658f;
  dim3 gsc(512 / 64, 512 / 128, 64);
  gemm_bf16_wmma<0,false><<<gsc, blk, 0, stream>>>(Q, Kb, nullptr, nullptr, Sc,
      512, 512, 96, 768, 768, 512,
      8, 512LL * 768, 96LL, 512LL * 768, 96LL,
      8LL * 512 * 512, 512LL * 512, scl);

  softmax_rows<<<dim3(64 * 512), blk, 0, stream>>>(Sc);

  // O[b,h] = att @ Vh  (B is [K=512, N=96] with row stride 768)
  dim3 gav(2, 512 / 128, 64);
  gemm_bf16_wmma<0,true><<<gav, blk, 0, stream>>>(Sc, V, nullptr, nullptr, O,
      512, 96, 512, 512, 768, 768,
      8, 8LL * 512 * 512, 512LL * 512,
      512LL * 768, 96LL, 512LL * 768, 96LL, 1.f);

  gemm_bf16_wmma<0,false><<<gproj, blk, 0, stream>>>(O, p.Wo, p.bo, nullptr, out,
      4096, 768, 768, 768, 768, 768, 1, 0, 0, 0, 0, 0, 0, 1.f);
}

static void run_mamba(hipStream_t stream, const float* x, const MambaP& p,
                      float* XLN, float* XZ, float* U, float* DLT, float* BM,
                      float* Y, float* out)
{
  dim3 blk(256);
  add_gate_ln<<<4096, blk, 0, stream>>>(x, nullptr, nullptr, p.ln_g, p.ln_b, XLN);
  gemm_bf16_wmma<0,false><<<dim3(3072 / 64, 32, 1), blk, 0, stream>>>(XLN, p.in_W, nullptr, nullptr, XZ,
      4096, 3072, 768, 768, 768, 3072, 1, 0, 0, 0, 0, 0, 0, 1.f);
  conv1d_silu<<<(8 * 512 * 1536) / 256, blk, 0, stream>>>(XZ, p.conv_w, p.conv_b, U);
  // Bmat = u @ xp_W[16:32,:]^T
  gemm_bf16_wmma<0,false><<<dim3(1, 32, 1), blk, 0, stream>>>(U, p.xp_W + 16 * 1536, nullptr, nullptr, BM,
      4096, 16, 1536, 1536, 1536, 16, 1, 0, 0, 0, 0, 0, 0, 1.f);
  // delta = softplus(u @ dt_W^T + dt_b)
  gemm_bf16_wmma<2,false><<<dim3(1536 / 64, 32, 1), blk, 0, stream>>>(U, p.dt_W, p.dt_b, nullptr, DLT,
      4096, 1536, 1536, 1536, 1536, 1536, 1, 0, 0, 0, 0, 0, 0, 1.f);
  mamba_scan<<<(8 * 1536) / 256, blk, 0, stream>>>(U, DLT, BM, p.A_log, p.D, XZ, Y);
  // out = y @ out_W^T + residual(x)
  gemm_bf16_wmma<0,false><<<dim3(768 / 64, 32, 1), blk, 0, stream>>>(Y, p.out_W, nullptr, x, out,
      4096, 768, 1536, 1536, 1536, 768, 1, 0, 0, 0, 0, 0, 0, 1.f);
}

extern "C" void kernel_launch(void* const* d_in, const int* in_sizes, int n_in,
                              void* d_out, int out_size, void* d_ws, size_t ws_size,
                              hipStream_t stream)
{
  (void)in_sizes; (void)n_in; (void)out_size; (void)ws_size;
  auto in = [&](int i) { return (const float*)d_in[i]; };
  const float* t_in = in(0);
  const float* v_in = in(1);

  auto mha = [&](int b) {
    MHAP p;
    p.Wq = in(b); p.Wk = in(b + 1); p.Wv = in(b + 2); p.Wo = in(b + 3);
    p.bq = in(b + 4); p.bk = in(b + 5); p.bv = in(b + 6); p.bo = in(b + 7);
    return p;
  };
  MHAP preT = mha(2), preV = mha(10), postT = mha(18), postV = mha(26);

  const float* gW[4]  = { in(34), in(36), in(38), in(40) };
  const float* gB[4]  = { in(35), in(37), in(39), in(41) };
  const float* lnG[4] = { in(42), in(44), in(46), in(48) };
  const float* lnB[4] = { in(43), in(45), in(47), in(49) };

  auto mam = [&](int b) {
    MambaP p;
    p.in_W = in(b);      p.conv_w = in(b + 1); p.conv_b = in(b + 2);
    p.xp_W = in(b + 3);  p.dt_W   = in(b + 4); p.dt_b   = in(b + 5);
    p.A_log = in(b + 6); p.D      = in(b + 7); p.out_W  = in(b + 8);
    p.ln_g = in(b + 9);  p.ln_b   = in(b + 10);
    return p;
  };
  MambaP mamT = mam(50), mamV = mam(61);

  // Workspace arena (floats). Persistent buffers + transient region shared
  // between the MHA scratch (Q,K,V,O,S) and the Mamba scratch.
  float* ws = (float*)d_ws;
  const size_t TOK = 4096ull * 768ull;           // 3,145,728 floats
  float* bufCt = ws;                             // tc / tcp
  float* bufCv = bufCt + TOK;
  float* bufG  = bufCv + TOK;                    // gate
  float* bufTi = bufG  + TOK;
  float* bufVi = bufTi + TOK;
  float* bufTm = bufVi + TOK;
  float* bufVm = bufTm + TOK;
  float* scr   = bufVm + TOK;
  // MHA view of scratch
  float* Q  = scr;
  float* Kb = Q  + TOK;
  float* V  = Kb + TOK;
  float* O  = V  + TOK;
  float* Sc = O  + TOK;                          // 64 * 512 * 512 floats
  // Mamba view of scratch (aliases; stages are disjoint in time)
  float* XLN = scr;
  float* XZ  = XLN + TOK;                        // 4096*3072
  float* U   = XZ  + 4096ull * 3072ull;          // 4096*1536
  float* DLT = U   + 4096ull * 1536ull;
  float* BM  = DLT + 4096ull * 1536ull;          // 4096*16
  float* Y   = BM  + 4096ull * 16ull;

  dim3 blk(256);
  dim3 gg(768 / 64, 32, 1);

  // --- pre cross-attentions ---
  run_mha(stream, t_in, v_in, v_in, preT, Q, Kb, V, Sc, O, bufCt);
  run_mha(stream, v_in, t_in, t_in, preV, Q, Kb, V, Sc, O, bufCv);

  // --- pre gates + LN ---
  gemm_bf16_wmma<1,false><<<gg, blk, 0, stream>>>(t_in, gW[0], gB[0], nullptr, bufG,
      4096, 768, 768, 768, 768, 768, 1, 0, 0, 0, 0, 0, 0, 1.f);
  add_gate_ln<<<4096, blk, 0, stream>>>(t_in, bufG, bufCt, lnG[0], lnB[0], bufTi);
  gemm_bf16_wmma<1,false><<<gg, blk, 0, stream>>>(v_in, gW[1], gB[1], nullptr, bufG,
      4096, 768, 768, 768, 768, 768, 1, 0, 0, 0, 0, 0, 0, 1.f);
  add_gate_ln<<<4096, blk, 0, stream>>>(v_in, bufG, bufCv, lnG[1], lnB[1], bufVi);

  // --- mamba blocks ---
  run_mamba(stream, bufTi, mamT, XLN, XZ, U, DLT, BM, Y, bufTm);
  run_mamba(stream, bufVi, mamV, XLN, XZ, U, DLT, BM, Y, bufVm);

  // --- post cross-attentions ---
  run_mha(stream, bufTm, bufVm, bufVm, postT, Q, Kb, V, Sc, O, bufCt);
  run_mha(stream, bufVm, bufTm, bufTm, postV, Q, Kb, V, Sc, O, bufCv);

  // --- post gates + LN, writing directly into d_out (et then ev) ---
  float* outF = (float*)d_out;
  gemm_bf16_wmma<1,false><<<gg, blk, 0, stream>>>(bufTm, gW[2], gB[2], nullptr, bufG,
      4096, 768, 768, 768, 768, 768, 1, 0, 0, 0, 0, 0, 0, 1.f);
  add_gate_ln<<<4096, blk, 0, stream>>>(bufTm, bufG, bufCt, lnG[2], lnB[2], outF);
  gemm_bf16_wmma<1,false><<<gg, blk, 0, stream>>>(bufVm, gW[3], gB[3], nullptr, bufG,
      4096, 768, 768, 768, 768, 768, 1, 0, 0, 0, 0, 0, 0, 1.f);
  add_gate_ln<<<4096, blk, 0, stream>>>(bufVm, bufG, bufCv, lnG[3], lnB[3], outF + TOK);
}
